// MultiHeadAttention_8040178778165
// MI455X (gfx1250) — compile-verified, matrix operand order checked by reading, not run on hardware
//
#include <hip/hip_runtime.h>
#include <hip/hip_bf16.h>
#include <math.h>

// ---------------------------------------------------------------------------
// MI455X (gfx1250) multi-head attention forward, bf16 WMMA + TDM staging.
// Shapes fixed by the reference: B=4, T=2048, C=1024, H=16, hs=64.
// ---------------------------------------------------------------------------

typedef __attribute__((ext_vector_type(16))) __bf16 v16bf;
typedef __attribute__((ext_vector_type(8)))  __bf16 v8bf;
typedef __attribute__((ext_vector_type(8)))  float  v8f;

typedef unsigned int v4u __attribute__((ext_vector_type(4)));
typedef int          v8i __attribute__((ext_vector_type(8)));
typedef int          v4i __attribute__((ext_vector_type(4)));

#define BB 4
#define TT 2048
#define CC 1024
#define HH 16
#define HS 64
#define HALFD 32               // hs/2 for RoPE
#define BT (BB * TT)           // 8192 rows

#if defined(__has_builtin)
#if __has_builtin(__builtin_amdgcn_tensor_load_to_lds) && \
    __has_builtin(__builtin_amdgcn_s_wait_tensorcnt)
#define HAVE_TDM 1
#endif
#endif
#ifndef HAVE_TDM
#define HAVE_TDM 0
#endif

union Frag16 { v16bf v; v8bf h[2]; };
union PackBF2 { __bf16 h[2]; unsigned u; };

// ---------------------------------------------------------------------------
// TDM: issue a 2D tile load (rows x cols bf16, row stride in elements) into
// LDS at byte offset lds_off.  Descriptor layout per CDNA5 ISA D# spec.
// ---------------------------------------------------------------------------
#if HAVE_TDM
__device__ inline void tdm_load_2d_bf16(unsigned lds_off, const void* gptr,
                                        unsigned rows, unsigned cols,
                                        unsigned stride_elems) {
  unsigned long long ga = (unsigned long long)(uintptr_t)gptr;
  v4u g0;
  g0[0] = 1u;                                       // count=1, user mode
  g0[1] = lds_off;                                  // lds_addr (bytes)
  g0[2] = (unsigned)(ga & 0xFFFFFFFFu);             // global_addr[31:0]
  g0[3] = (unsigned)((ga >> 32) & 0x01FFFFFFu) | 0x80000000u;  // addr hi | type=2
  v8i g1;
  g1[0] = (int)(1u << 16);                          // data_size=1 (2 bytes)
  g1[1] = (int)((cols & 0xFFFFu) << 16);            // tensor_dim0[15:0] @bit48
  g1[2] = (int)(((cols >> 16) & 0xFFFFu) |          // tensor_dim0[31:16]
                ((rows & 0xFFFFu) << 16));          // tensor_dim1[15:0]
  g1[3] = (int)(((rows >> 16) & 0xFFFFu) |          // tensor_dim1[31:16]
                ((cols & 0xFFFFu) << 16));          // tile_dim0 = cols
  g1[4] = (int)(rows & 0xFFFFu);                    // tile_dim1 = rows, tile_dim2=0
  g1[5] = (int)stride_elems;                        // tensor_dim0_stride[31:0]
  g1[6] = 0;                                        // stride hi, dim1_stride lo
  g1[7] = 0;
  v4i z4 = {0, 0, 0, 0};
#if __clang_major__ >= 23
  v8i z8 = {0, 0, 0, 0, 0, 0, 0, 0};
  __builtin_amdgcn_tensor_load_to_lds(g0, g1, z4, z4, z8, 0);
#else
  __builtin_amdgcn_tensor_load_to_lds(g0, g1, z4, z4, 0);
#endif
}
#endif

// ---------------------------------------------------------------------------
// f32 -> bf16 conversion (memory-bound, trivial vs 23.3 TB/s)
// ---------------------------------------------------------------------------
__global__ void cvt_f32_bf16(const float* __restrict__ src,
                             __bf16* __restrict__ dst, int n) {
  int i = blockIdx.x * blockDim.x + threadIdx.x;
  if (i < n) dst[i] = (__bf16)src[i];
}

// ---------------------------------------------------------------------------
// Tiled GEMM:  Out[M,N] = A[M,K](bf16) @ B[K,N](bf16) + bias[N](f32)
// Block tile 128x64, K-step 32.  8 waves, each wave owns a 2x2 grid of 16x16
// C tiles (A and B fragments each reused twice -> 4 WMMAs per wave per step).
// B staged transposed in LDS so fragments are contiguous 16B chunks per lane.
// ---------------------------------------------------------------------------
template <bool OUT_BF16>
__global__ void gemm_bf16_wmma(const __bf16* __restrict__ A,
                               const __bf16* __restrict__ Bm,
                               const float* __restrict__ bias,
                               void* __restrict__ Out,
                               int M, int N, int K) {
  __shared__ __align__(16) __bf16 ldsA[128][32];
  __shared__ __align__(16) __bf16 ldsBt[64][32];   // [n][k]

  const int tid  = threadIdx.x;
  const int wv   = tid >> 5;
  const int lane = tid & 31;
  const int hv   = lane >> 4;      // half-wave selects K sub-range
  const int l16  = lane & 15;

  const int mBase = blockIdx.y * 128;
  const int nBase = blockIdx.x * 64;
  const int mi0 = (wv >> 1) * 2;   // wave's first row tile (0,2,4,6)
  const int ni0 = (wv & 1) * 2;    // wave's first col tile (0 or 2)

  v8f acc00 = {}, acc01 = {}, acc10 = {}, acc11 = {};

  for (int k0 = 0; k0 < K; k0 += 32) {
    { // A tile 128x32, 16 bf16 (32B) per thread
      int r = tid >> 1, c = (tid & 1) * 16;
      const uint4* g = (const uint4*)&A[(size_t)(mBase + r) * K + k0 + c];
      uint4* s = (uint4*)&ldsA[r][c];
      s[0] = g[0]; s[1] = g[1];
    }
    { // B tile 32x64, transposed into [n][k]
      int kk = tid >> 3, nc = (tid & 7) * 8;
      v8bf row = *(const v8bf*)&Bm[(size_t)(k0 + kk) * N + nBase + nc];
#pragma unroll
      for (int j = 0; j < 8; ++j) ldsBt[nc + j][kk] = row[j];
    }
    __syncthreads();

    Frag16 FA0, FA1, FB0, FB1;
    FA0.h[0] = *(const v8bf*)&ldsA[mi0 * 16 + l16][hv * 8];
    FA0.h[1] = *(const v8bf*)&ldsA[mi0 * 16 + l16][16 + hv * 8];
    FA1.h[0] = *(const v8bf*)&ldsA[(mi0 + 1) * 16 + l16][hv * 8];
    FA1.h[1] = *(const v8bf*)&ldsA[(mi0 + 1) * 16 + l16][16 + hv * 8];
    FB0.h[0] = *(const v8bf*)&ldsBt[ni0 * 16 + l16][hv * 8];
    FB0.h[1] = *(const v8bf*)&ldsBt[ni0 * 16 + l16][16 + hv * 8];
    FB1.h[0] = *(const v8bf*)&ldsBt[(ni0 + 1) * 16 + l16][hv * 8];
    FB1.h[1] = *(const v8bf*)&ldsBt[(ni0 + 1) * 16 + l16][16 + hv * 8];

    acc00 = __builtin_amdgcn_wmma_f32_16x16x32_bf16(
        false, FA0.v, false, FB0.v, (short)0, acc00, false, false);
    acc01 = __builtin_amdgcn_wmma_f32_16x16x32_bf16(
        false, FA0.v, false, FB1.v, (short)0, acc01, false, false);
    acc10 = __builtin_amdgcn_wmma_f32_16x16x32_bf16(
        false, FA1.v, false, FB0.v, (short)0, acc10, false, false);
    acc11 = __builtin_amdgcn_wmma_f32_16x16x32_bf16(
        false, FA1.v, false, FB1.v, (short)0, acc11, false, false);
    __syncthreads();
  }

  const int col0 = nBase + ni0 * 16 + l16;
  const float b0 = bias[col0];
  const float b1 = bias[col0 + 16];
#pragma unroll
  for (int r = 0; r < 8; ++r) {
    const int row0 = mBase + mi0 * 16 + hv * 8 + r;  // C/D layout: reg r = M row
    const int row1 = row0 + 16;
    if (OUT_BF16) {
      __bf16* O = (__bf16*)Out;
      O[(size_t)row0 * N + col0]      = (__bf16)(acc00[r] + b0);
      O[(size_t)row0 * N + col0 + 16] = (__bf16)(acc01[r] + b1);
      O[(size_t)row1 * N + col0]      = (__bf16)(acc10[r] + b0);
      O[(size_t)row1 * N + col0 + 16] = (__bf16)(acc11[r] + b1);
    } else {
      float* O = (float*)Out;
      O[(size_t)row0 * N + col0]      = acc00[r] + b0;
      O[(size_t)row0 * N + col0 + 16] = acc01[r] + b1;
      O[(size_t)row1 * N + col0]      = acc10[r] + b0;
      O[(size_t)row1 * N + col0 + 16] = acc11[r] + b1;
    }
  }
}

// ---------------------------------------------------------------------------
// RoPE on q and k inside the bf16 qkv buffer [B*T][3C].
// ---------------------------------------------------------------------------
__global__ void rope_kernel(__bf16* __restrict__ qkv) {
  int idx = blockIdx.x * blockDim.x + threadIdx.x;
  int i = idx & (HALFD - 1);
  int h = (idx >> 5) & (HH - 1);
  int t = (idx >> 9) & (TT - 1);
  int b = idx >> 20;
  if (b >= BB) return;

  size_t row = (size_t)(b * TT + t) * (3 * CC);
  size_t qo  = row + h * HS + 2 * i;
  size_t ko  = qo + CC;

  float theta = __powf(10000.f, -(float)i / (float)HALFD);
  float ang = (float)t * theta;
  float s, c;
  __sincosf(ang, &s, &c);

  float qe = (float)qkv[qo], qd = (float)qkv[qo + 1];
  qkv[qo]     = (__bf16)(qe * c - qd * s);
  qkv[qo + 1] = (__bf16)(qe * s + qd * c);
  float ke = (float)qkv[ko], kd = (float)qkv[ko + 1];
  qkv[ko]     = (__bf16)(ke * c - kd * s);
  qkv[ko + 1] = (__bf16)(ke * s + kd * c);
}

// ---------------------------------------------------------------------------
// Flash-style causal attention, 64-query x 64-key tiles.
// Q/K tiles staged by the Tensor Data Mover (wave 0 issues the D#); V staged
// transposed by all waves concurrently; online softmax in LDS; O in VGPRs.
// Dynamic LDS with fixed byte offsets so the D# lds_addr is known.
// ---------------------------------------------------------------------------
#define OFF_Q   0u          //  64*64 bf16 =  8192
#define OFF_K   8192u       //  64*64 bf16 =  8192
#define OFF_VT  16384u      //  64*64 bf16 =  8192 ([d][key])
#define OFF_S   24576u      //  64*64 f32  = 16384
#define OFF_P   40960u      //  64*64 bf16 =  8192
#define OFF_M   49152u
#define OFF_L   49408u
#define OFF_AL  49664u
#define SMEM_ATTN 49920u

__global__ void attn_kernel(const __bf16* __restrict__ qkv,
                            __bf16* __restrict__ outA) {
  extern __shared__ __align__(16) char smem[];
  __bf16 (*sQ)[64]  = (__bf16(*)[64])(smem + OFF_Q);
  __bf16 (*sK)[64]  = (__bf16(*)[64])(smem + OFF_K);
  __bf16 (*sVt)[64] = (__bf16(*)[64])(smem + OFF_VT);   // [d][key]
  float  (*sS)[64]  = (float(*)[64])(smem + OFF_S);
  __bf16 (*sP)[64]  = (__bf16(*)[64])(smem + OFF_P);
  float* sM  = (float*)(smem + OFF_M);
  float* sL  = (float*)(smem + OFF_L);
  float* sAl = (float*)(smem + OFF_AL);

  const int qTile = blockIdx.x;
  const int h = blockIdx.y, b = blockIdx.z;
  const int qBase = qTile * 64;
  const int tid  = threadIdx.x;
  const int wv   = tid >> 5;
  const int lane = tid & 31;
  const int hv   = lane >> 4;
  const int l16  = lane & 15;

  const size_t pitch = 3 * CC;
  const size_t bRow  = (size_t)b * TT;
  const int dOff = h * HS;
  const float norm = 0.125f;             // 1/sqrt(64)

  // ---- Q tile (TDM if available, else staged loads) ----
#if HAVE_TDM
  if (wv == 0) {
    tdm_load_2d_bf16(OFF_Q, &qkv[(bRow + qBase) * pitch + dOff], 64, 64, 3 * CC);
    __builtin_amdgcn_s_wait_tensorcnt(0);
  }
#else
  {
    int r = tid >> 2, c = (tid & 3) * 16;
    const uint4* g = (const uint4*)&qkv[(bRow + qBase + r) * pitch + dOff + c];
    uint4* s = (uint4*)&sQ[r][c];
    s[0] = g[0]; s[1] = g[1];
  }
#endif
  if (tid < 64) { sM[tid] = -INFINITY; sL[tid] = 0.f; }
  __syncthreads();

  const int smi = wv >> 1, snj0 = (wv & 1) * 2;   // S: 4x4 tiles, 2 per wave
  const int omi = wv >> 1, oni0 = (wv & 1) * 2;   // O: 4x4 tiles, 2 per wave
  v8f o0 = {}, o1 = {};

  const int nTiles = qBase / 64 + 1;              // causal bound
  for (int j = 0; j < nTiles; ++j) {
    const int jBase = j * 64;

    // ---- K tile 64x64 via TDM (wave 0); V tile transposed by everyone ----
#if HAVE_TDM
    if (wv == 0)
      tdm_load_2d_bf16(OFF_K, &qkv[(bRow + jBase) * pitch + CC + dOff],
                       64, 64, 3 * CC);
#else
    {
      int r = tid >> 2, c = (tid & 3) * 16;
      const uint4* g =
          (const uint4*)&qkv[(bRow + jBase + r) * pitch + CC + dOff + c];
      uint4* s = (uint4*)&sK[r][c];
      s[0] = g[0]; s[1] = g[1];
    }
#endif
    { // V: 32 row-pairs x 8 col-groups; packed b32 stores into sVt[d][key]
      int r0 = (tid >> 3) * 2, c = (tid & 7) * 8;
      v8bf ra = *(const v8bf*)&qkv[(bRow + jBase + r0) * pitch + 2 * CC + dOff + c];
      v8bf rb = *(const v8bf*)&qkv[(bRow + jBase + r0 + 1) * pitch + 2 * CC + dOff + c];
#pragma unroll
      for (int e = 0; e < 8; ++e) {
        PackBF2 p; p.h[0] = ra[e]; p.h[1] = rb[e];
        *(unsigned*)&sVt[c + e][r0] = p.u;
      }
    }
    // prefetch next key tile while this one is consumed
    if (j + 1 < nTiles)
      __builtin_prefetch(&qkv[(bRow + jBase + 64) * pitch + CC + dOff], 0, 1);
#if HAVE_TDM
    if (wv == 0) __builtin_amdgcn_s_wait_tensorcnt(0);
#endif
    __syncthreads();

    // ---- S tiles = Q @ K^T (contract d=64 -> 2 chained WMMAs per tile) ----
    v8f s0 = {}, s1 = {};
#pragma unroll
    for (int kk = 0; kk < 64; kk += 32) {
      Frag16 FQ, FK0, FK1;
      FQ.h[0]  = *(const v8bf*)&sQ[smi * 16 + l16][kk + hv * 8];
      FQ.h[1]  = *(const v8bf*)&sQ[smi * 16 + l16][kk + 16 + hv * 8];
      FK0.h[0] = *(const v8bf*)&sK[snj0 * 16 + l16][kk + hv * 8];
      FK0.h[1] = *(const v8bf*)&sK[snj0 * 16 + l16][kk + 16 + hv * 8];
      FK1.h[0] = *(const v8bf*)&sK[(snj0 + 1) * 16 + l16][kk + hv * 8];
      FK1.h[1] = *(const v8bf*)&sK[(snj0 + 1) * 16 + l16][kk + 16 + hv * 8];
      s0 = __builtin_amdgcn_wmma_f32_16x16x32_bf16(
          false, FQ.v, false, FK0.v, (short)0, s0, false, false);
      s1 = __builtin_amdgcn_wmma_f32_16x16x32_bf16(
          false, FQ.v, false, FK1.v, (short)0, s1, false, false);
    }
    { // scale + causal mask + spill both tiles to LDS
      const int kc0 = jBase + snj0 * 16 + l16;
#pragma unroll
      for (int r = 0; r < 8; ++r) {
        const int qrow = qBase + smi * 16 + hv * 8 + r;
        float v0 = s0[r] * norm, v1 = s1[r] * norm;
        if (kc0 > qrow)      v0 = -INFINITY;
        if (kc0 + 16 > qrow) v1 = -INFINITY;
        sS[smi * 16 + hv * 8 + r][snj0 * 16 + l16]      = v0;
        sS[smi * 16 + hv * 8 + r][(snj0 + 1) * 16 + l16] = v1;
      }
    }
    __syncthreads();

    // ---- online softmax: one thread per query row ----
    if (tid < 64) {
      float m = sM[tid];
      float mNew = m;
#pragma unroll 8
      for (int c = 0; c < 64; ++c) mNew = fmaxf(mNew, sS[tid][c]);
      const float alpha = __expf(m - mNew);
      float sum = 0.f;
#pragma unroll 8
      for (int c = 0; c < 64; ++c) {
        const float p = __expf(sS[tid][c] - mNew);
        sP[tid][c] = (__bf16)p;
        sum += p;
      }
      sM[tid]  = mNew;
      sL[tid]  = sL[tid] * alpha + sum;
      sAl[tid] = alpha;
    }
    __syncthreads();

    // ---- O = O*alpha + P @ V (contract keys=64 -> 2 WMMAs per tile) ----
#pragma unroll
    for (int r = 0; r < 8; ++r) {
      const float a = sAl[omi * 16 + hv * 8 + r];
      o0[r] *= a; o1[r] *= a;
    }
#pragma unroll
    for (int kk = 0; kk < 64; kk += 32) {
      Frag16 FP, FV0, FV1;
      FP.h[0]  = *(const v8bf*)&sP[omi * 16 + l16][kk + hv * 8];
      FP.h[1]  = *(const v8bf*)&sP[omi * 16 + l16][kk + 16 + hv * 8];
      FV0.h[0] = *(const v8bf*)&sVt[oni0 * 16 + l16][kk + hv * 8];
      FV0.h[1] = *(const v8bf*)&sVt[oni0 * 16 + l16][kk + 16 + hv * 8];
      FV1.h[0] = *(const v8bf*)&sVt[(oni0 + 1) * 16 + l16][kk + hv * 8];
      FV1.h[1] = *(const v8bf*)&sVt[(oni0 + 1) * 16 + l16][kk + 16 + hv * 8];
      o0 = __builtin_amdgcn_wmma_f32_16x16x32_bf16(
          false, FP.v, false, FV0.v, (short)0, o0, false, false);
      o1 = __builtin_amdgcn_wmma_f32_16x16x32_bf16(
          false, FP.v, false, FV1.v, (short)0, o1, false, false);
    }
    __syncthreads();
  }

  // ---- finalize: O /= l, write bf16 [B*T][C] (heads interleaved == ref) ----
#pragma unroll
  for (int r = 0; r < 8; ++r) {
    const int rowT = omi * 16 + hv * 8 + r;
    const float inv = 1.f / sL[rowT];
    const size_t gr = (bRow + qBase + rowT) * CC + dOff;
    outA[gr + oni0 * 16 + l16]       = (__bf16)(o0[r] * inv);
    outA[gr + (oni0 + 1) * 16 + l16] = (__bf16)(o1[r] * inv);
  }
}

// ---------------------------------------------------------------------------
// Host-side orchestration (graph-capture safe: launches only).
// ---------------------------------------------------------------------------
extern "C" void kernel_launch(void* const* d_in, const int* in_sizes, int n_in,
                              void* d_out, int out_size, void* d_ws,
                              size_t ws_size, hipStream_t stream) {
  const float* x    = (const float*)d_in[0];
  const float* Wqkv = (const float*)d_in[1];
  const float* bqkv = (const float*)d_in[2];
  const float* Wout = (const float*)d_in[3];
  const float* bout = (const float*)d_in[4];
  float* out = (float*)d_out;

  // Workspace layout (bf16 buffers), ~88 MB total.
  char* w = (char*)d_ws;
  __bf16* xb    = (__bf16*)w;  w += (size_t)BT * CC * 2;       // 16 MB
  __bf16* wqkvb = (__bf16*)w;  w += (size_t)CC * 3 * CC * 2;   //  6 MB
  __bf16* woutb = (__bf16*)w;  w += (size_t)CC * CC * 2;       //  2 MB
  __bf16* qkvb  = (__bf16*)w;  w += (size_t)BT * 3 * CC * 2;   // 48 MB
  __bf16* attb  = (__bf16*)w;                                  // 16 MB

  const int nX = BT * CC, nWq = CC * 3 * CC, nWo = CC * CC;
  cvt_f32_bf16<<<(nX + 255) / 256, 256, 0, stream>>>(x, xb, nX);
  cvt_f32_bf16<<<(nWq + 255) / 256, 256, 0, stream>>>(Wqkv, wqkvb, nWq);
  cvt_f32_bf16<<<(nWo + 255) / 256, 256, 0, stream>>>(Wout, woutb, nWo);

  // qkv = x @ Wqkv + bqkv   (M=8192, N=3072, K=1024) -> bf16
  gemm_bf16_wmma<true><<<dim3(3 * CC / 64, BT / 128), 256, 0, stream>>>(
      xb, wqkvb, bqkv, (void*)qkvb, BT, 3 * CC, CC);

  // RoPE on q, k
  rope_kernel<<<(BB * TT * HH * HALFD) / 256, 256, 0, stream>>>(qkvb);

  // causal flash attention -> attb bf16 [B*T][C]
  attn_kernel<<<dim3(TT / 64, HH, BB), 256, SMEM_ATTN, stream>>>(qkvb, attb);

  // out = attb @ Wout + bout  (M=8192, N=1024, K=1024) -> f32 d_out
  gemm_bf16_wmma<false><<<dim3(CC / 64, BT / 128), 256, 0, stream>>>(
      attb, woutb, bout, (void*)out, BT, CC, CC);
}
